// SelfAttention3D_8521215115500
// MI455X (gfx1250) — compile-verified, matrix operand order checked by reading, not run on hardware
//
#include <hip/hip_runtime.h>

typedef __bf16 bf16_t;
typedef __attribute__((ext_vector_type(16))) __bf16 v16bf;
typedef __attribute__((ext_vector_type(8)))  float  v8f;

constexpr int Bn = 4;      // batch
constexpr int Cc = 256;    // channels
constexpr int CQ = 32;     // q/k dim (C/8)
constexpr int Nn = 4096;   // tokens (16*16*16)
constexpr int JT = 64;     // attention j-chunk (keys/values per staging round)

// K index inside a 16-bit A/B fragment, per CDNA5 ISA 7.12.2:
// lane half 0 holds K {0..7,16..23}, half 1 holds K {8..15,24..31},
// VGPR i holds the pair (kk, kk+1).
__device__ __forceinline__ int frag_k(int i, int half) {
    return ((i < 4) ? 0 : 16) + half * 8 + 2 * (i & 3);
}

// ---------------------------------------------------------------------------
// Kernel 0: one-shot fp32 -> bf16 weight conversion (Wq|Wk|Wv concatenated).
// ---------------------------------------------------------------------------
__global__ __launch_bounds__(256) void wprep_kernel(
    const float* __restrict__ Wq, const float* __restrict__ Wk,
    const float* __restrict__ Wv, bf16_t* __restrict__ Wb)
{
    const int nq = CQ * Cc;            // 8192
    const int nv = Cc * Cc;            // 65536
    const int total = 2 * nq + nv;     // 81920
    for (int i = blockIdx.x * 256 + threadIdx.x; i < total; i += gridDim.x * 256) {
        float v;
        if (i < nq)            v = Wq[i];
        else if (i < 2 * nq)   v = Wk[i - nq];
        else                   v = Wv[i - 2 * nq];
        Wb[i] = (bf16_t)v;
    }
}

// ---------------------------------------------------------------------------
// Kernel 1: QKV projections (1x1 conv == token x channel GEMM).
// Block: 128 threads (4 waves), one (batch, 16-token) tile per block.
// Q -> [B][N][32] (token-major), K -> [B][N][32] (token-major),
// V -> [B][256][N] (channel-major: contiguous stores, B-fragment friendly).
// ---------------------------------------------------------------------------
__global__ __launch_bounds__(128) void qkv_kernel(
    const float* __restrict__ x,
    const bf16_t* __restrict__ Wb,      // bf16 Wq|Wk|Wv
    const float* __restrict__ bq, const float* __restrict__ bk,
    const float* __restrict__ bv,
    bf16_t* __restrict__ Qb, bf16_t* __restrict__ Kb, bf16_t* __restrict__ Vb)
{
    __shared__ bf16_t sx[16][Cc + 8];   // x^T tile: [token][channel], bf16

    const int blk = blockIdx.x;
    const int b   = blk / (Nn / 16);
    const int n0  = (blk % (Nn / 16)) * 16;
    const int t    = threadIdx.x;
    const int wave = t >> 5;
    const int lane = t & 31;
    const int half = lane >> 4;
    const int lm   = lane & 15;

    // stage x[b][c][n0..n0+15] -> sx[n][c]
    for (int idx = t; idx < 16 * Cc; idx += 128) {
        int c = idx >> 4;
        int n = idx & 15;
        sx[n][c] = (bf16_t)x[((size_t)b * Cc + c) * Nn + n0 + n];
    }
    __syncthreads();

    const bf16_t* Wqb = Wb;
    const bf16_t* Wkb = Wb + CQ * Cc;
    const bf16_t* Wvb = Wb + 2 * CQ * Cc;

    // 20 output tiles of 16 channels: 0-1 = Q, 2-3 = K, 4-19 = V
    for (int tile = wave; tile < 20; tile += 4) {
        const bf16_t* Wmat; const float* bias; bf16_t* dst; int o0; bool isV;
        if (tile < 2)      { Wmat = Wqb; bias = bq; dst = Qb; o0 = tile * 16;       isV = false; }
        else if (tile < 4) { Wmat = Wkb; bias = bk; dst = Kb; o0 = (tile - 2) * 16; isV = false; }
        else               { Wmat = Wvb; bias = bv; dst = Vb; o0 = (tile - 4) * 16; isV = true;  }

        v8f acc = {};
        const bf16_t* wrow = &Wmat[(size_t)(o0 + lm) * Cc];
        for (int k0 = 0; k0 < Cc; k0 += 32) {
            v16bf a, bm;
#pragma unroll
            for (int i = 0; i < 8; ++i) {
                const int kk = frag_k(i, half);
                a[2*i]   = sx[lm][k0 + kk];       // A[m][k] = x^T[token][chan]
                a[2*i+1] = sx[lm][k0 + kk + 1];
                bm[2*i]   = wrow[k0 + kk];        // B[k][n] = W[o0+n][k]
                bm[2*i+1] = wrow[k0 + kk + 1];
            }
            acc = __builtin_amdgcn_wmma_f32_16x16x32_bf16(
                false, a, false, bm, (short)0, acc, false, false);
        }
        const float bia = bias[o0 + lm];
        if (isV) {
            // channel-major: lane's 8 rows are 8 consecutive tokens -> 16B store
            union { bf16_t h[8]; uint4 u; } tmp;
#pragma unroll
            for (int r = 0; r < 8; ++r) tmp.h[r] = (bf16_t)(acc[r] + bia);
            *(uint4*)&dst[((size_t)b * Cc + o0 + lm) * Nn + n0 + 8 * half] = tmp.u;
        } else {
#pragma unroll
            for (int r = 0; r < 8; ++r) {
                const int m = r + 8 * half;   // token row within tile (C/D layout)
                dst[((size_t)b * Nn + n0 + m) * CQ + (o0 + lm)] =
                    (bf16_t)(acc[r] + bia);
            }
        }
    }
}

// ---------------------------------------------------------------------------
// Kernel 2: flash attention + epilogue. Block: 256 threads (8 waves) covers
// 128 query rows of one batch; each wave owns 16 rows x all 256 channels.
// 64-token chunks: 36 WMMAs per softmax/rescale round.
// ---------------------------------------------------------------------------
__global__ __launch_bounds__(256) void attn_kernel(
    const bf16_t* __restrict__ Qb, const bf16_t* __restrict__ Kb,
    const bf16_t* __restrict__ Vb,
    const float* __restrict__ x, const float* __restrict__ gamma,
    float* __restrict__ out)
{
    __shared__ bf16_t sk[JT][40];          // K chunk: [token j][dim d]
    __shared__ bf16_t sv[Cc][JT + 8];      // V chunk: [channel c][token j]
    __shared__ bf16_t sp[8][16][JT + 8];   // per-wave P transpose scratch

    const int t    = threadIdx.x;
    const int wave = t >> 5;
    const int lane = t & 31;
    const int half = lane >> 4;
    const int lm   = lane & 15;
    const int b    = blockIdx.x >> 5;                       // 32 blocks / batch
    const int i0   = (blockIdx.x & 31) * 128 + wave * 16;   // first query row

    // Preload this wave's Q A-fragment (16 rows x K=32, reused all chunks).
    v16bf qa;
    {
        const bf16_t* qrow = &Qb[((size_t)b * Nn + i0 + lm) * CQ];
#pragma unroll
        for (int i = 0; i < 8; ++i) {
            const int kk = frag_k(i, half);
            qa[2*i]   = qrow[kk];
            qa[2*i+1] = qrow[kk + 1];
        }
    }

    v8f oacc[16];
#pragma unroll
    for (int ct = 0; ct < 16; ++ct) oacc[ct] = {};
    float mrow[8], lrow[8];
#pragma unroll
    for (int r = 0; r < 8; ++r) { mrow[r] = -3.0e38f; lrow[r] = 0.f; }

    for (int j0 = 0; j0 < Nn; j0 += JT) {
        // Prefetch next chunk's K/V rows (global_prefetch_b8).
        if (j0 + JT < Nn) {
            __builtin_prefetch(&Vb[((size_t)b * Cc + t) * Nn + j0 + JT], 0, 0);
            if (t < JT)
                __builtin_prefetch(&Kb[((size_t)b * Nn + j0 + JT + t) * CQ], 0, 0);
        }
        // Cooperative staging, 8-byte vectorized.
        // K chunk: 64 tokens x 32 dims = 2048 bf16 = 512 uint2
#pragma unroll
        for (int it = 0; it < 2; ++it) {
            const int idx = t + it * 256;
            const int j   = idx >> 3;
            const int d4  = (idx & 7) * 4;
            *(uint2*)&sk[j][d4] =
                *(const uint2*)&Kb[((size_t)b * Nn + j0 + j) * CQ + d4];
        }
        // V chunk: 256 channels x 64 tokens = 16384 bf16 = 4096 uint2
#pragma unroll
        for (int it = 0; it < 16; ++it) {
            const int idx = t + it * 256;
            const int c   = idx >> 4;
            const int j4  = (idx & 15) * 4;
            *(uint2*)&sv[c][j4] =
                *(const uint2*)&Vb[((size_t)b * Cc + c) * Nn + j0 + j4];
        }
        __syncthreads();

        // S = Q * K^T : four 16x16 subtiles (j cols 0-15..48-63), K-dim 32.
        v8f s[4];
#pragma unroll
        for (int jt = 0; jt < 4; ++jt) {
            v16bf bk;
#pragma unroll
            for (int i = 0; i < 8; ++i) {
                const int kk = frag_k(i, half);
                bk[2*i]   = sk[jt * 16 + lm][kk];
                bk[2*i+1] = sk[jt * 16 + lm][kk + 1];
            }
            v8f z = {};
            s[jt] = __builtin_amdgcn_wmma_f32_16x16x32_bf16(
                false, qa, false, bk, (short)0, z, false, false);
        }

        // Online softmax. Row m = r + 8*half lives in one 16-lane half.
        float scale[8];
#pragma unroll
        for (int r = 0; r < 8; ++r) {
            float rv = fmaxf(fmaxf(s[0][r], s[1][r]), fmaxf(s[2][r], s[3][r]));
            for (int msk = 1; msk < 16; msk <<= 1)
                rv = fmaxf(rv, __shfl_xor(rv, msk, 32));
            const float mnew = fmaxf(mrow[r], rv);
            scale[r] = __expf(mrow[r] - mnew);
            float rs = 0.f;
            const int m = r + 8 * half;
#pragma unroll
            for (int jt = 0; jt < 4; ++jt) {
                const float p = __expf(s[jt][r] - mnew);
                rs += p;
                sp[wave][m][jt * 16 + lm] = (bf16_t)p;   // C/D layout stash
            }
            for (int msk = 1; msk < 16; msk <<= 1)
                rs += __shfl_xor(rs, msk, 32);
            lrow[r] = lrow[r] * scale[r] + rs;
            mrow[r] = mnew;
        }
#pragma unroll
        for (int ct = 0; ct < 16; ++ct)
#pragma unroll
            for (int r = 0; r < 8; ++r) oacc[ct][r] *= scale[r];

        // Reload P as two A-fragments (16 rows x K=32 of j each).
        v16bf pa0, pa1;
#pragma unroll
        for (int i = 0; i < 8; ++i) {
            const int kk = frag_k(i, half);
            pa0[2*i]   = sp[wave][lm][kk];
            pa0[2*i+1] = sp[wave][lm][kk + 1];
            pa1[2*i]   = sp[wave][lm][32 + kk];
            pa1[2*i+1] = sp[wave][lm][32 + kk + 1];
        }

        // O += P * V across 16 channel tiles, two K-steps each (32 WMMAs).
#pragma unroll
        for (int ct = 0; ct < 16; ++ct) {
            v16bf bv0, bv1;
#pragma unroll
            for (int i = 0; i < 8; ++i) {
                const int kk = frag_k(i, half);
                bv0[2*i]   = sv[ct * 16 + lm][kk];
                bv0[2*i+1] = sv[ct * 16 + lm][kk + 1];
                bv1[2*i]   = sv[ct * 16 + lm][32 + kk];
                bv1[2*i+1] = sv[ct * 16 + lm][32 + kk + 1];
            }
            oacc[ct] = __builtin_amdgcn_wmma_f32_16x16x32_bf16(
                false, pa0, false, bv0, (short)0, oacc[ct], false, false);
            oacc[ct] = __builtin_amdgcn_wmma_f32_16x16x32_bf16(
                false, pa1, false, bv1, (short)0, oacc[ct], false, false);
        }
        __syncthreads();
    }

    // Epilogue: out = gamma * (O / l) + x, fp32.
    const float g = gamma[0];
#pragma unroll
    for (int ct = 0; ct < 16; ++ct) {
#pragma unroll
        for (int r = 0; r < 8; ++r) {
            const int c = ct * 16 + lm;
            const int i = i0 + r + 8 * half;
            const size_t idx = ((size_t)b * Cc + c) * Nn + i;
            out[idx] = g * (oacc[ct][r] / lrow[r]) + x[idx];
        }
    }
}

// ---------------------------------------------------------------------------
extern "C" void kernel_launch(void* const* d_in, const int* in_sizes, int n_in,
                              void* d_out, int out_size, void* d_ws, size_t ws_size,
                              hipStream_t stream) {
    const float* x     = (const float*)d_in[0];
    const float* Wq    = (const float*)d_in[1];
    const float* bq    = (const float*)d_in[2];
    const float* Wk    = (const float*)d_in[3];
    const float* bk    = (const float*)d_in[4];
    const float* Wv    = (const float*)d_in[5];
    const float* bv    = (const float*)d_in[6];
    const float* gamma = (const float*)d_in[7];

    bf16_t* Qb = (bf16_t*)d_ws;                       // [B][N][32]
    bf16_t* Kb = Qb + (size_t)Bn * Nn * CQ;           // [B][N][32]
    bf16_t* Vb = Kb + (size_t)Bn * Nn * CQ;           // [B][256][N] channel-major
    bf16_t* Wb = Vb + (size_t)Bn * Nn * Cc;           // bf16 Wq|Wk|Wv (81920)

    wprep_kernel<<<40, 256, 0, stream>>>(Wq, Wk, Wv, Wb);
    qkv_kernel<<<Bn * (Nn / 16), 128, 0, stream>>>(
        x, Wb, bq, bk, bv, Qb, Kb, Vb);
    attn_kernel<<<Bn * (Nn / 128), 256, 0, stream>>>(
        Qb, Kb, Vb, x, gamma, (float*)d_out);
}
